// ClusterProtoNetwork_15006615733014
// MI455X (gfx1250) — compile-verified, matrix operand order checked by reading, not run on hardware
//
#include <hip/hip_runtime.h>
#include <math.h>

// ---------------------------------------------------------------------------
// Problem constants (from reference)
// ---------------------------------------------------------------------------
#define N_WAYC     16
#define N_SUPPORTC 256
#define N_QUERYC   512
#define D_INC      4096
#define D_EMBC     1024
#define NCLUST     5
#define MAX_ITERC  100

#define M_TOTAL (N_WAYC * (N_SUPPORTC + N_QUERYC))   // 12288 rows (4096 support + 8192 query)
#define M_SUP   (N_WAYC * N_SUPPORTC)                // 4096

typedef __attribute__((ext_vector_type(16))) __bf16 v16bf;
typedef __attribute__((ext_vector_type(8)))  float  v8f;

union FragB16 {
    v16bf v;
    uint4 q[2];
};

__device__ __forceinline__ unsigned short f2bf(float f) {
    union { float f; unsigned int u; } x; x.f = f;
    unsigned int u = x.u;
    // round-to-nearest-even bf16 truncation
    return (unsigned short)((u + 0x7FFFu + ((u >> 16) & 1u)) >> 16);
}

__device__ __forceinline__ unsigned long long pack4bf(float4 v) {
    unsigned int lo = (unsigned int)f2bf(v.x) | ((unsigned int)f2bf(v.y) << 16);
    unsigned int hi = (unsigned int)f2bf(v.z) | ((unsigned int)f2bf(v.w) << 16);
    return (unsigned long long)lo | ((unsigned long long)hi << 32);
}

// ---------------------------------------------------------------------------
// CDNA5 async global->LDS copy (ASYNCcnt path), with portable fallback.
// Builtin signature (from hipcc diagnostic): (v4i AS1*, v4i AS3*, imm, imm)
// ---------------------------------------------------------------------------
#if __has_builtin(__builtin_amdgcn_global_load_async_to_lds_b128) && \
    __has_builtin(__builtin_amdgcn_s_wait_asynccnt)
#define USE_ASYNC_LDS 1
typedef int v4i __attribute__((vector_size(16)));
typedef __attribute__((address_space(1))) v4i gv4i;   // global-AS pointee
typedef __attribute__((address_space(3))) v4i lv4i;   // LDS-AS pointee
#else
#define USE_ASYNC_LDS 0
#endif

__device__ __forceinline__ void cp_g2l_b128(const unsigned short* g, unsigned short* l) {
#if USE_ASYNC_LDS
    __builtin_amdgcn_global_load_async_to_lds_b128((gv4i*)g, (lv4i*)l, 0, 0);
#else
    *(uint4*)l = *(const uint4*)g;
#endif
}

__device__ __forceinline__ void wait_async0() {
#if USE_ASYNC_LDS
    __builtin_amdgcn_s_wait_asynccnt(0);
#endif
}

// ---------------------------------------------------------------------------
// Kernel 0a: X = concat(support, query) f32 -> bf16 (one-shot; makes the
//            96 MB bf16 X + 8 MB Wt L2-resident for the whole GEMM)
// ---------------------------------------------------------------------------
__global__ __launch_bounds__(256) void xcvt_kernel(
    const float* __restrict__ S, const float* __restrict__ Q,
    unsigned short* __restrict__ Xbf)
{
    const size_t SUP = (size_t)M_SUP * D_INC;
    size_t idx = ((size_t)blockIdx.x * 256 + threadIdx.x) * 4;
    const float* src = (idx < SUP) ? (S + idx) : (Q + (idx - SUP));
    float4 v = *(const float4*)src;
    *(unsigned long long*)(Xbf + idx) = pack4bf(v);
}

// ---------------------------------------------------------------------------
// Kernel 0b: W [4096 k][1024 n] f32 -> Wt [1024 n][4096 k] bf16 (transpose+cvt)
// ---------------------------------------------------------------------------
__global__ __launch_bounds__(256) void wt_transpose_kernel(
    const float* __restrict__ W, unsigned short* __restrict__ Wt)
{
    __shared__ float tile[32][33];
    const int tid = threadIdx.x;
    const int tx = tid & 31;
    const int ty = tid >> 5;
    const int k0 = blockIdx.x * 32;
    const int n0 = blockIdx.y * 32;

#pragma unroll
    for (int i = 0; i < 32; i += 8)
        tile[ty + i][tx] = W[(size_t)(k0 + ty + i) * D_EMBC + n0 + tx];
    __syncthreads();
#pragma unroll
    for (int i = 0; i < 32; i += 8)
        Wt[(size_t)(n0 + ty + i) * D_INC + k0 + tx] = f2bf(tile[tx][ty + i]);
}

// ---------------------------------------------------------------------------
// Kernel 1: encoder GEMM  E[m][n] = X[m][k] * W[k][n] + b[n]
//   M=12288, K=4096, N=1024. bf16 WMMA, f32 accumulate.
//   Block tile 128x128, BK=64, 256 threads (8 waves), wave tile 32x64.
//   Double-buffered LDS; tiles staged with GLOBAL_LOAD_ASYNC_TO_LDS_B128.
// ---------------------------------------------------------------------------
#define BK 64
#define LDSTRIDE 72   // bf16/row: 144 B = 16*9 (b128-aligned, conflict-free frags)
#define KITERS (D_INC / BK)

__global__ __launch_bounds__(256) void encode_gemm_kernel(
    const unsigned short* __restrict__ Xbf, const unsigned short* __restrict__ Wt,
    const float* __restrict__ bias, float* __restrict__ emb)
{
    __shared__ unsigned short As[2][128 * LDSTRIDE];   // 2 x 18 KB
    __shared__ unsigned short Bs[2][128 * LDSTRIDE];   // 2 x 18 KB

    const int tid  = threadIdx.x;
    const int wave = tid >> 5;
    const int lane = tid & 31;
    const int wm   = (wave & 3) * 32;   // wave M offset within block tile
    const int wn   = (wave >> 2) * 64;  // wave N offset within block tile
    const int row0 = blockIdx.y * 128;  // global M base
    const int col0 = blockIdx.x * 128;  // global N base

    const unsigned short* Xa = Xbf + (size_t)row0 * D_INC;
    const unsigned short* Wa = Wt + (size_t)col0 * D_INC;

    // staging chunk owned by this thread (4 b128 chunks per tile each for A/B)
    const int cr = tid >> 3;            // 0..31 base row, +32 per chunk
    const int ck = (tid & 7) << 3;      // k element offset 0..56

    v8f acc[2][4];
#pragma unroll
    for (int mt = 0; mt < 2; ++mt)
#pragma unroll
        for (int nt = 0; nt < 4; ++nt)
#pragma unroll
            for (int r = 0; r < 8; ++r)
                acc[mt][nt][r] = 0.0f;

    const int lm  = lane & 15;            // row (A) / col (B) within 16
    const int lk8 = (lane >> 4) << 3;     // 0 or 8: which K-half the lane owns

    // ---- prologue: stage tile 0 into buffer 0 ----
#pragma unroll
    for (int i = 0; i < 4; ++i) {
        int r = cr + i * 32;
        cp_g2l_b128(Xa + (size_t)r * D_INC + ck, &As[0][r * LDSTRIDE + ck]);
        cp_g2l_b128(Wa + (size_t)r * D_INC + ck, &Bs[0][r * LDSTRIDE + ck]);
    }

    for (int it = 0; it < KITERS; ++it) {
        const int cur = it & 1;
        wait_async0();          // current buffer's copies complete (own wave)
        __syncthreads();        // all waves' copies visible; prev compute done

        if (it + 1 < KITERS) {  // prefetch next tile into the other buffer
            const int nxt = cur ^ 1;
            const int kn  = (it + 1) * BK;
#pragma unroll
            for (int i = 0; i < 4; ++i) {
                int r = cr + i * 32;
                cp_g2l_b128(Xa + (size_t)r * D_INC + kn + ck, &As[nxt][r * LDSTRIDE + ck]);
                cp_g2l_b128(Wa + (size_t)r * D_INC + kn + ck, &Bs[nxt][r * LDSTRIDE + ck]);
            }
        }

        // ---- two K=32 WMMA steps over the staged tiles ----
#pragma unroll
        for (int ks = 0; ks < BK; ks += 32) {
            FragB16 a[2], b[4];
#pragma unroll
            for (int mt = 0; mt < 2; ++mt) {
                const unsigned short* p = &As[cur][(wm + mt * 16 + lm) * LDSTRIDE + ks + lk8];
                a[mt].q[0] = *(const uint4*)p;
                a[mt].q[1] = *(const uint4*)(p + 16);
            }
#pragma unroll
            for (int nt = 0; nt < 4; ++nt) {
                const unsigned short* p = &Bs[cur][(wn + nt * 16 + lm) * LDSTRIDE + ks + lk8];
                b[nt].q[0] = *(const uint4*)p;
                b[nt].q[1] = *(const uint4*)(p + 16);
            }
#pragma unroll
            for (int mt = 0; mt < 2; ++mt)
#pragma unroll
                for (int nt = 0; nt < 4; ++nt)
                    acc[mt][nt] = __builtin_amdgcn_wmma_f32_16x16x32_bf16(
                        false, a[mt].v, false, b[nt].v,
                        (short)0, acc[mt][nt], false, false);
        }
    }

    // ---- epilogue: add bias, store f32 embeddings ----
    const int mbase = (lane >> 4) * 8;   // lanes 16-31 hold M+8 rows
#pragma unroll
    for (int nt = 0; nt < 4; ++nt) {
        int n = col0 + wn + nt * 16 + lm;
        float bv = bias[n];
#pragma unroll
        for (int mt = 0; mt < 2; ++mt) {
#pragma unroll
            for (int r = 0; r < 8; ++r) {
                int m = row0 + wm + mt * 16 + r + mbase;
                emb[(size_t)m * D_EMBC + n] = acc[mt][nt][r] + bv;
            }
        }
    }
}

// ---------------------------------------------------------------------------
// Kernel 2: per-class Lloyd k-means (k=5, 100 iters) on support embeddings.
// ---------------------------------------------------------------------------
__global__ __launch_bounds__(256) void kmeans_kernel(
    const float* __restrict__ emb, float* __restrict__ protos)
{
    __shared__ float cent[NCLUST][D_EMBC];   // 20 KB
    __shared__ float cnorm[NCLUST];
    __shared__ int   assign_s[N_SUPPORTC];
    __shared__ int   cnt[NCLUST];

    const int c   = blockIdx.x;
    const int tid = threadIdx.x;
    const float* base = emb + (size_t)c * N_SUPPORTC * D_EMBC;
    const int d0 = tid * 4;

#pragma unroll
    for (int j = 0; j < NCLUST; ++j) {
        int idx = (17 + j * 101) & (N_SUPPORTC - 1);
        for (int d = tid; d < D_EMBC; d += 256)
            cent[j][d] = base[(size_t)idx * D_EMBC + d];
    }
    if (tid < NCLUST) cnorm[tid] = 0.0f;
    __syncthreads();
#pragma unroll
    for (int j = 0; j < NCLUST; ++j) {
        float p = 0.0f;
#pragma unroll
        for (int i = 0; i < 4; ++i) { float v = cent[j][d0 + i]; p += v * v; }
        atomicAdd(&cnorm[j], p);
    }
    __syncthreads();

    for (int it = 0; it < MAX_ITERC; ++it) {
        // ---- assignment: thread = point; argmin_j (|c_j|^2 - 2 q.c_j) ----
        const float* q = base + (size_t)tid * D_EMBC;
        float dot[NCLUST] = {0.f, 0.f, 0.f, 0.f, 0.f};
        for (int d = 0; d < D_EMBC; d += 4) {
            float4 v = *(const float4*)(q + d);
#pragma unroll
            for (int j = 0; j < NCLUST; ++j) {
                dot[j] += v.x * cent[j][d]     + v.y * cent[j][d + 1]
                        + v.z * cent[j][d + 2] + v.w * cent[j][d + 3];
            }
        }
        int best = 0;
        float bd = cnorm[0] - 2.0f * dot[0];
#pragma unroll
        for (int j = 1; j < NCLUST; ++j) {
            float dj = cnorm[j] - 2.0f * dot[j];
            if (dj < bd) { bd = dj; best = j; }
        }
        assign_s[tid] = best;
        if (tid < NCLUST) cnt[tid] = 0;
        __syncthreads();
        atomicAdd(&cnt[best], 1);
        if (tid < NCLUST) cnorm[tid] = 0.0f;   // all reads done pre-barrier
        __syncthreads();

        // ---- segment sums: thread owns dims d0..d0+3, loop points (coalesced) ----
        float s[NCLUST][4];
#pragma unroll
        for (int j = 0; j < NCLUST; ++j)
#pragma unroll
            for (int i = 0; i < 4; ++i) s[j][i] = 0.0f;

        for (int p = 0; p < N_SUPPORTC; ++p) {
            int a = __builtin_amdgcn_readfirstlane(assign_s[p]);  // wave-uniform
            float4 v = *(const float4*)(base + (size_t)p * D_EMBC + d0);
#pragma unroll
            for (int j = 0; j < NCLUST; ++j) {
                if (a == j) {
                    s[j][0] += v.x; s[j][1] += v.y; s[j][2] += v.z; s[j][3] += v.w;
                }
            }
        }

        // ---- centroid update + cnorm rebuild (empty cluster keeps old) ----
#pragma unroll
        for (int j = 0; j < NCLUST; ++j) {
            int cj = cnt[j];
            float inv = (cj > 0) ? (1.0f / (float)cj) : 0.0f;
            float p = 0.0f;
#pragma unroll
            for (int i = 0; i < 4; ++i) {
                float v = (cj > 0) ? s[j][i] * inv : cent[j][d0 + i];
                cent[j][d0 + i] = v;
                p += v * v;
            }
            atomicAdd(&cnorm[j], p);
        }
        __syncthreads();
    }

    for (int d = tid; d < D_EMBC; d += 256) {
        float sm = 0.0f;
#pragma unroll
        for (int j = 0; j < NCLUST; ++j) sm += cent[j][d];
        protos[(size_t)c * D_EMBC + d] = sm * 0.2f;
    }
}

// ---------------------------------------------------------------------------
// Kernel 3: logits = -cdist(query_emb, class_protos)
// ---------------------------------------------------------------------------
__global__ __launch_bounds__(256) void logits_kernel(
    const float* __restrict__ emb, const float* __restrict__ protos,
    float* __restrict__ out)
{
    __shared__ float P[8][D_EMBC];   // 32 KB per chunk
    const int tid = threadIdx.x;
    const int gid = blockIdx.x * 256 + tid;
    const float* q = emb + (size_t)(M_SUP + gid) * D_EMBC;

    for (int ch = 0; ch < 2; ++ch) {
        for (int i = tid; i < 8 * D_EMBC; i += 256)
            ((float*)P)[i] = protos[ch * 8 * D_EMBC + i];
        __syncthreads();

        float d2[8] = {0.f, 0.f, 0.f, 0.f, 0.f, 0.f, 0.f, 0.f};
        for (int d = 0; d < D_EMBC; d += 4) {
            float4 v = *(const float4*)(q + d);
#pragma unroll
            for (int cc = 0; cc < 8; ++cc) {
                float dx = v.x - P[cc][d];
                float dy = v.y - P[cc][d + 1];
                float dz = v.z - P[cc][d + 2];
                float dw = v.w - P[cc][d + 3];
                d2[cc] += dx * dx + dy * dy + dz * dz + dw * dw;
            }
        }
#pragma unroll
        for (int cc = 0; cc < 8; ++cc)
            out[(size_t)gid * N_WAYC + ch * 8 + cc] = -sqrtf(fmaxf(d2[cc], 0.0f));
        __syncthreads();
    }
}

// ---------------------------------------------------------------------------
// Launch
// ---------------------------------------------------------------------------
extern "C" void kernel_launch(void* const* d_in, const int* in_sizes, int n_in,
                              void* d_out, int out_size, void* d_ws, size_t ws_size,
                              hipStream_t stream) {
    const float* support = (const float*)d_in[0];   // [16,256,4096]
    const float* query   = (const float*)d_in[1];   // [16,512,4096]
    const float* W       = (const float*)d_in[2];   // [4096,1024]
    const float* bias    = (const float*)d_in[3];   // [1024]
    float* out = (float*)d_out;                     // [16,512,16]

    char* ws = (char*)d_ws;
    const size_t XBF_BYTES = (size_t)M_TOTAL * D_INC * sizeof(unsigned short); // 96 MB
    const size_t WT_BYTES  = (size_t)D_EMBC * D_INC * sizeof(unsigned short);  // 8 MB
    const size_t EMB_BYTES = (size_t)M_TOTAL * D_EMBC * sizeof(float);         // 48 MB
    unsigned short* Xbf = (unsigned short*)ws;
    unsigned short* Wt  = (unsigned short*)(ws + XBF_BYTES);
    float* emb          = (float*)(ws + XBF_BYTES + WT_BYTES);
    float* protos       = (float*)(ws + XBF_BYTES + WT_BYTES + EMB_BYTES);

    // 0) one-shot conversions: X -> bf16, W -> bf16 transposed (L2-resident)
    xcvt_kernel<<<(M_TOTAL * (D_INC / 4)) / 256, 256, 0, stream>>>(support, query, Xbf);
    wt_transpose_kernel<<<dim3(D_INC / 32, D_EMBC / 32), 256, 0, stream>>>(W, Wt);

    // 1) encoder GEMM: N-tiles fastest varying -> A tiles shared through L2
    encode_gemm_kernel<<<dim3(D_EMBC / 128, M_TOTAL / 128), 256, 0, stream>>>(
        Xbf, Wt, bias, emb);

    // 2) per-class k-means (k=5, 100 iterations)
    kmeans_kernel<<<N_WAYC, 256, 0, stream>>>(emb, protos);

    // 3) logits
    logits_kernel<<<(N_WAYC * N_QUERYC) / 256, 256, 0, stream>>>(emb, protos, out);
}
